// attention_80796924773020
// MI455X (gfx1250) — matrix operand order repaired and checked by reading, not compiled, not run on hardware
//
#include <hip/hip_runtime.h>

// Problem constants (from reference): B=4, H=16, S=2048, D=64
constexpr int Bc = 4, Hc = 16, Sc = 2048, Dc = 64;
constexpr int BQ = 128;       // query rows per block (8 waves x 16 rows)
constexpr int KB = 32;        // keys per inner iteration
constexpr int NW = 8;         // waves per block
constexpr int NIT = Sc / KB;  // 64 iterations

typedef __bf16 v16bf __attribute__((ext_vector_type(16)));
typedef __bf16 v8bf  __attribute__((ext_vector_type(8)));
typedef float  v8f   __attribute__((ext_vector_type(8)));
typedef float  v4f   __attribute__((ext_vector_type(4)));

union PAu { v16bf v; v8bf h[2]; };

// max without LLVM's maxnum canonicalization overhead
__device__ __forceinline__ float fmax_raw(float a, float b) {
    float d;
    asm("v_max_num_f32 %0, %1, %2" : "=v"(d) : "v"(a), "v"(b));
    return d;
}

// Max-reduce + broadcast across each 16-lane group, one fused DPP-max per
// step: dst = max(dpp(src0), src1). Sequence: xor1, xor2, then mirrors
// (valid once quads/octets are uniform). Pure VALU, no LDS.
__device__ __forceinline__ float rowmax16(float x) {
    asm("v_max_num_f32_dpp %0, %0, %0 quad_perm:[1,0,3,2] row_mask:0xf bank_mask:0xf bound_ctrl:1\n\t"
        "v_max_num_f32_dpp %0, %0, %0 quad_perm:[2,3,0,1] row_mask:0xf bank_mask:0xf bound_ctrl:1\n\t"
        "v_max_num_f32_dpp %0, %0, %0 row_half_mirror row_mask:0xf bank_mask:0xf bound_ctrl:1\n\t"
        "v_max_num_f32_dpp %0, %0, %0 row_mirror row_mask:0xf bank_mask:0xf bound_ctrl:1"
        : "+v"(x));
    return x;
}

// Pack 4 bool bytes (0/1) of a dword into a nibble.
__device__ __forceinline__ unsigned int pack4(unsigned int x) {
    return (x | (x >> 7) | (x >> 14) | (x >> 21)) & 0xFu;
}

__global__ __launch_bounds__(256) void fa_wmma_kernel(
    const float* __restrict__ q,
    const float* __restrict__ k,
    const float* __restrict__ v,
    const unsigned char* __restrict__ mask,
    float* __restrict__ out)
{
    __shared__ __align__(16) __bf16 lds_k[2][KB * Dc];         // [key][d]   2x4KB
    __shared__ __align__(16) __bf16 lds_vt[2][Dc * KB];        // [d][key]   2x4KB
    __shared__ __align__(4)  unsigned short lds_mb[2][BQ * 2]; // mask bits  2x512B
    __shared__ __align__(16) __bf16 lds_p[NW][16 * KB];        // per-wave P 8KB

    const int tid  = threadIdx.x;
    const int w    = tid >> 5;
    const int lane = tid & 31;
    const int hf   = lane >> 4;   // 0: lanes 0-15, 1: lanes 16-31
    const int ln   = lane & 15;
    const int bh   = blockIdx.y;
    const int b    = bh / Hc;
    const int h    = bh % Hc;
    const int qbase = blockIdx.x * BQ;

    const float scale = 0.125f;            // 1/sqrt(64)
    const float L2E   = 1.44269504088896f; // log2(e)

    const size_t bh_off = ((size_t)(b * Hc + h)) * Sc * Dc;
    const float* qp = q + bh_off;
    const float* kp = k + bh_off;
    const float* vp = v + bh_off;
    float*       op = out + bh_off;
    const unsigned char* mp = mask + (size_t)b * Sc * Sc;

    // ---- preload Q tile into WMMA A-layout registers (pre-scaled) ----
    v16bf qa[2];
    const int qrow  = qbase + w * 16 + ln;
    const int koff8 = hf * 16;
    #pragma unroll
    for (int a = 0; a < 2; ++a) {
        const float* r0 = qp + (size_t)qrow * Dc + a * 32 + koff8;
        v4f t0 = *(const v4f*)(r0);
        v4f t1 = *(const v4f*)(r0 + 4);
        v4f t2 = *(const v4f*)(r0 + 8);
        v4f t3 = *(const v4f*)(r0 + 12);
        #pragma unroll
        for (int e = 0; e < 4; ++e) {
            qa[a][e]      = (__bf16)(t0[e] * scale);
            qa[a][e + 4]  = (__bf16)(t1[e] * scale);
            qa[a][e + 8]  = (__bf16)(t2[e] * scale);
            qa[a][e + 12] = (__bf16)(t3[e] * scale);
        }
    }

    // constants for the reduction-by-WMMA trick
    v16bf ones;
    #pragma unroll
    for (int e = 0; e < 16; ++e) ones[e] = (__bf16)1.0f;
    v8f vzero;
    #pragma unroll
    for (int r = 0; r < 8; ++r) vzero[r] = 0.0f;

    // softmax running state: C-matrix VGPR slot r holds row M = r + 8*hf
    float mrow[8], lrow[8];
    v8f acc[4];
    #pragma unroll
    for (int r = 0; r < 8; ++r) { mrow[r] = -1e30f; lrow[r] = 0.0f; }
    #pragma unroll
    for (int dc = 0; dc < 4; ++dc)
        #pragma unroll
        for (int r = 0; r < 8; ++r) acc[dc][r] = 0.0f;

    // cooperative-load coordinates (256 threads)
    const int key_c   = tid >> 3;         // K: key row (0..31), coalesced global
    const int d0_c    = (tid & 7) * 8;    // K: 8 consecutive d
    const int vkey_c  = tid & 31;         // V: key = lane -> spread LDS banks
    const int vd0_c   = (tid >> 5) * 8;   // V: 8 d per thread, per-wave range
    const int mrow_c  = tid >> 1;         // mask row (0..127)
    const int mhalf_c = tid & 1;          // mask 16-col half

    auto load_tile = [&](int kbi, int buf) {
        const int kbase = kbi * KB;
        // K tile: coalesced global b128 loads -> packed v8bf LDS store
        const float* ks = kp + (size_t)(kbase + key_c) * Dc + d0_c;
        v4f a0 = *(const v4f*)(ks);
        v4f a1 = *(const v4f*)(ks + 4);
        v8bf tk;
        #pragma unroll
        for (int e = 0; e < 4; ++e) {
            tk[e]     = (__bf16)a0[e];
            tk[e + 4] = (__bf16)a1[e];
        }
        *(v8bf*)&lds_k[buf][key_c * Dc + d0_c] = tk;
        // V tile transposed: key = lane spreads LDS banks on the scatter
        const float* vs = vp + (size_t)(kbase + vkey_c) * Dc + vd0_c;
        v4f b0 = *(const v4f*)(vs);
        v4f b1 = *(const v4f*)(vs + 4);
        #pragma unroll
        for (int e = 0; e < 4; ++e) {
            lds_vt[buf][(vd0_c + e) * KB + vkey_c]     = (__bf16)b0[e];
            lds_vt[buf][(vd0_c + 4 + e) * KB + vkey_c] = (__bf16)b1[e];
        }
        // mask tile: 16 bytes -> 16 bits per thread; fold "col 0 always true"
        uint4 mt = *(const uint4*)(mp + (size_t)(qbase + mrow_c) * Sc + kbase + mhalf_c * 16);
        unsigned int bits = pack4(mt.x) | (pack4(mt.y) << 4) |
                            (pack4(mt.z) << 8) | (pack4(mt.w) << 12);
        if (kbi == 0 && mhalf_c == 0) bits |= 1u;
        lds_mb[buf][mrow_c * 2 + mhalf_c] = (unsigned short)bits;
        if (kbi + 1 < NIT) {
            __builtin_prefetch(ks + KB * Dc, 0, 1);
            __builtin_prefetch(vs + KB * Dc, 0, 1);
        }
    };

    load_tile(0, 0);

    for (int kb = 0; kb < NIT; ++kb) {
        const int buf = kb & 1;
        __syncthreads();                    // tile[buf] ready; tile[buf^1] free
        if (kb + 1 < NIT) load_tile(kb + 1, buf ^ 1);

        // -------- S = Q K^T (two 16-key tiles, two K=32 d-chunks) --------
        v8f sc[2];
        #pragma unroll
        for (int c = 0; c < 2; ++c) {
            v16bf bk0 = *(const v16bf*)&lds_k[buf][(c * 16 + ln) * Dc + 0 * 32 + hf * 16];
            v16bf bk1 = *(const v16bf*)&lds_k[buf][(c * 16 + ln) * Dc + 1 * 32 + hf * 16];
            sc[c] = __builtin_amdgcn_wmma_f32_16x16x32_bf16(
                        false, qa[0], false, bk0, (short)0, vzero, false, false);
            sc[c] = __builtin_amdgcn_wmma_f32_16x16x32_bf16(
                        false, qa[1], false, bk1, (short)0, sc[c], false, false);
        }

        // -------- running max from RAW scores (mask only shrinks p later;
        //          any shift >= true masked max is valid for softmax) --------
        float corr[8];
        #pragma unroll
        for (int r = 0; r < 8; ++r) {
            const float m = rowmax16(fmax_raw(sc[0][r], sc[1][r]));
            const float mnew = fmax_raw(mrow[r], m);
            corr[r] = __expf(mrow[r] - mnew);
            mrow[r] = mnew;
            lrow[r] *= corr[r];
            #pragma unroll
            for (int dc = 0; dc < 4; ++dc) acc[dc][r] *= corr[r];
        }

        // -------- p = exp2(s*log2e - m*log2e) masked by sign-extended bit ----
        #pragma unroll
        for (int r = 0; r < 8; ++r) {
            const unsigned int bits =
                *(const unsigned int*)&lds_mb[buf][(w * 16 + r + hf * 8) * 2];
            const float ml2 = mrow[r] * L2E;
            #pragma unroll
            for (int c = 0; c < 2; ++c) {
                const float p = exp2f(__builtin_fmaf(sc[c][r], L2E, -ml2));
                const int sel = __builtin_amdgcn_sbfe((int)bits, c * 16 + ln, 1);
                const float pmsk = __int_as_float(__float_as_int(p) & sel);
                lds_p[w][(r + hf * 8) * KB + c * 16 + ln] = (__bf16)pmsk;
            }
        }
        PAu pa;
        {
            const __bf16* pr = &lds_p[w][ln * KB + koff8];
            pa.h[0] = *(const v8bf*)(pr);
            pa.h[1] = *(const v8bf*)(pr + 8);
        }

        // -------- row sums via WMMA against all-ones B --------
        v8f lsum = __builtin_amdgcn_wmma_f32_16x16x32_bf16(
                       false, pa.v, false, ones, (short)0, vzero, false, false);
        #pragma unroll
        for (int r = 0; r < 8; ++r) lrow[r] += lsum[r];

        // -------- O += P V (4 d-chunks of 16) --------
        #pragma unroll
        for (int dc = 0; dc < 4; ++dc) {
            v16bf bv = *(const v16bf*)&lds_vt[buf][(dc * 16 + ln) * KB + hf * 16];
            acc[dc] = __builtin_amdgcn_wmma_f32_16x16x32_bf16(
                          false, pa.v, false, bv, (short)0, acc[dc], false, false);
        }
    }

    // -------- epilogue: normalize and store --------
    #pragma unroll
    for (int r = 0; r < 8; ++r) {
        const int row = qbase + w * 16 + r + hf * 8;
        const float inv = 1.0f / lrow[r];
        #pragma unroll
        for (int dc = 0; dc < 4; ++dc)
            op[(size_t)row * Dc + dc * 16 + ln] = acc[dc][r] * inv;
    }
}

extern "C" void kernel_launch(void* const* d_in, const int* in_sizes, int n_in,
                              void* d_out, int out_size, void* d_ws, size_t ws_size,
                              hipStream_t stream) {
    const float* q = (const float*)d_in[0];
    const float* k = (const float*)d_in[1];
    const float* v = (const float*)d_in[2];
    const unsigned char* mask = (const unsigned char*)d_in[3];
    float* out = (float*)d_out;
    (void)in_sizes; (void)n_in; (void)out_size; (void)d_ws; (void)ws_size;

    dim3 grid(Sc / BQ, Bc * Hc);   // 16 x 64 blocks
    dim3 block(256);               // 8 waves (wave32)
    fa_wmma_kernel<<<grid, block, 0, stream>>>(q, k, v, mask, out);
}
